// GlobalCCMambaBlock_19129784336934
// MI455X (gfx1250) — matrix operand order repaired and checked by reading, not compiled, not run on hardware
//
#include <hip/hip_runtime.h>
#include <hip/hip_bf16.h>

typedef float v8f __attribute__((ext_vector_type(8)));
typedef _Float16 v16h __attribute__((ext_vector_type(16)));
typedef _Float16 v8h  __attribute__((ext_vector_type(8)));

union frag16 { v16h v; v8h h[2]; };

#define CHUNK 128

__device__ __forceinline__ v8f wmma_f16(v16h a, v16h b, v8f c) {
  // D(16x16 f32) = A(16x32 f16) * B(32x16 f16) + C
  return __builtin_amdgcn_wmma_f32_16x16x32_f16(
      /*neg_a=*/false, a, /*neg_b=*/false, b,
      /*c_mod=*/(short)0, c, /*reuse_a=*/false, /*reuse_b=*/false);
}

__device__ __forceinline__ float silu_f(float v) {
  return v / (1.f + __expf(-v));
}

// gfx1250 async DMA: global memory -> LDS, 16B per issue, tracked by ASYNCcnt.
__device__ __forceinline__ void async_copy_b128(void* lds_dst, const void* gsrc) {
  // LDS aperture lives in the high 32 bits of the flat address; the low 32 bits
  // are exactly the workgroup-relative LDS byte offset the instruction needs.
  unsigned l = (unsigned)(size_t)lds_dst;
  asm volatile("global_load_async_to_lds_b128 %0, %1, off"
               :: "v"(l), "v"(gsrc) : "memory");
}
__device__ __forceinline__ void async_wait0() {
  asm volatile("s_wait_asynccnt 0x0" ::: "memory");
}

// ---------------- 1) scores = x @ W_score + b ----------------
__global__ void k_score(const float* __restrict__ x, const float* __restrict__ Wsc,
                        const float* __restrict__ bsc, float* __restrict__ scores, int n) {
  int lane = threadIdx.x & 31;
  int row  = blockIdx.x * 8 + (threadIdx.x >> 5);
  if (row >= n) return;
  float acc = 0.f;
#pragma unroll
  for (int k = lane; k < 128; k += 32) acc += x[(size_t)row * 128 + k] * Wsc[k];
#pragma unroll
  for (int off = 16; off; off >>= 1) acc += __shfl_xor(acc, off, 32);
  if (lane == 0) scores[row] = acc + bsc[0];
}

// ---------------- 2) stable rank-count argsort ----------------
__global__ void k_rank(const float* __restrict__ scores, int* __restrict__ idx, int n) {
  __shared__ float tile[1024];
  int t = blockIdx.x * blockDim.x + threadIdx.x;
  float my = (t < n) ? scores[t] : 0.f;
  int rank = 0;
  for (int base = 0; base < n; base += 1024) {
    int m = min(1024, n - base);
    __syncthreads();
    for (int j = threadIdx.x; j < m; j += blockDim.x) tile[j] = scores[base + j];
    __syncthreads();
    if (t < n) {
      for (int j = 0; j < m; ++j) {
        float s = tile[j];
        rank += (s < my) || (s == my && (base + j) < t);
      }
    }
  }
  if (t < n) idx[rank] = t;
}

// ---------------- 3) gather + LayerNorm -> f16 activations ----------------
__global__ void k_ln(const float* __restrict__ x, const int* __restrict__ idx,
                     const float* __restrict__ gamma, const float* __restrict__ beta,
                     _Float16* __restrict__ xs, int n) {
  int lane = threadIdx.x & 31;
  int row  = blockIdx.x * 8 + (threadIdx.x >> 5);
  if (row >= n) return;
  int g = idx[row];
  float v[4];
  float s = 0.f, s2 = 0.f;
#pragma unroll
  for (int i = 0; i < 4; i++) {
    float t = x[(size_t)g * 128 + lane + i * 32];
    v[i] = t; s += t; s2 += t * t;
  }
#pragma unroll
  for (int off = 16; off; off >>= 1) {
    s  += __shfl_xor(s, off, 32);
    s2 += __shfl_xor(s2, off, 32);
  }
  float mu  = s * (1.f / 128.f);
  float var = s2 * (1.f / 128.f) - mu * mu;
  float inv = rsqrtf(var + 1e-5f);
#pragma unroll
  for (int i = 0; i < 4; i++) {
    int k = lane + i * 32;
    xs[(size_t)row * 128 + k] = (_Float16)((v[i] - mu) * inv * gamma[k] + beta[k]);
  }
}

// ---------------- weight convert + transpose to K-contiguous f16 ----------------
// W_in (128 x 512) -> Wt (512 x 128): Wt[ncol*128 + k] = W[k*512 + ncol]
__global__ void k_cvt_in(const float* __restrict__ W, _Float16* __restrict__ Wt) {
  int id = blockIdx.x * 256 + threadIdx.x;  // 65536
  int k = id & 127, ncol = id >> 7;
  Wt[id] = (_Float16)W[(size_t)k * 512 + ncol];
}
// W_out (256 x 128) -> Wt (128 x 256): Wt[ncol*256 + k] = W[k*128 + ncol]
__global__ void k_cvt_out(const float* __restrict__ W, _Float16* __restrict__ Wt) {
  int id = blockIdx.x * 256 + threadIdx.x;  // 32768
  int k = id & 255, ncol = id >> 8;
  Wt[id] = (_Float16)W[(size_t)k * 128 + ncol];
}

// ---------------- 4) xz = xs @ W_in  (N x 128 @ 128 x 512), f16 WMMA, f32 acc ----------------
__global__ void k_gemm_in(const _Float16* __restrict__ A /* N x 128 */,
                          const _Float16* __restrict__ Bt /* 512 x 128, K-contig */,
                          float* __restrict__ C, int n) {
  int lane = threadIdx.x & 31;
  int w    = threadIdx.x >> 5;
  int row0 = blockIdx.x * 16;
  int col0 = w * 64;
  int m   = lane & 15;
  int kbA = (lane >> 4) << 3;   // 0 or 8
  int kbB = (lane >> 4) << 3;   // 0 or 8

  v8f zero = {0.f, 0.f, 0.f, 0.f, 0.f, 0.f, 0.f, 0.f};
  v8f acc[4] = {zero, zero, zero, zero};
  v8h zh = {0, 0, 0, 0, 0, 0, 0, 0};

  int r = row0 + m;
#pragma unroll
  for (int k0 = 0; k0 < 128; k0 += 32) {
    frag16 a;
    if (r < n) {
      const _Float16* ap = A + (size_t)r * 128 + k0 + kbA;
      a.h[0] = *(const v8h*)(ap);        // K = k0+kbA .. +7
      a.h[1] = *(const v8h*)(ap + 16);   // K = k0+16+kbA .. +7
    } else {
      a.h[0] = zh; a.h[1] = zh;
    }
#pragma unroll
    for (int g = 0; g < 4; ++g) {
      const _Float16* bp = Bt + (size_t)(col0 + g * 16 + m) * 128 + k0 + kbB;
      frag16 b;
      b.h[0] = *(const v8h*)(bp);        // K = k0+kbB .. +7
      b.h[1] = *(const v8h*)(bp + 16);   // K = k0+16+kbB .. +7
      acc[g] = wmma_f16(a.v, b.v, acc[g]);
    }
  }
  int rbase = row0 + ((lane >> 4) << 3);
#pragma unroll
  for (int v = 0; v < 8; v++) {
    int rr = rbase + v;
    if (rr >= n) continue;
#pragma unroll
    for (int g = 0; g < 4; g++) {
      int cc = col0 + g * 16 + m;
      C[(size_t)rr * 512 + cc] = acc[g][v];
    }
  }
}

// ---------------- 5) depthwise causal conv(4) + SiLU ----------------
__global__ void k_conv(const float* __restrict__ xz, const float* __restrict__ cw,
                       const float* __restrict__ cb, float* __restrict__ uc, int n) {
  int id = blockIdx.x * blockDim.x + threadIdx.x;
  if (id >= n * 256) return;
  int t = id >> 8, c = id & 255;
  float acc = cb[c];
#pragma unroll
  for (int k = 0; k < 4; k++) {
    int tk = t - 3 + k;
    if (tk >= 0) acc += xz[(size_t)tk * 512 + c] * cw[c * 4 + k];
  }
  uc[(size_t)t * 256 + c] = silu_f(acc);
}

// ---------------- 6) dbc = uc @ W_xp ; delta = softplus(dbc[:,:8]@W_dt + b) ; B,C ----------------
__global__ void k_proj(const float* __restrict__ uc, const float* __restrict__ Wxp,
                       const float* __restrict__ Wdt, const float* __restrict__ bdt,
                       float* __restrict__ delta, float* __restrict__ Bm,
                       float* __restrict__ Cm, int n) {
  __shared__ float s_uc[8][256];
  __shared__ float s_dbc[8][40];
  int lane = threadIdx.x & 31;
  int w    = threadIdx.x >> 5;
  int row  = blockIdx.x * 8 + w;
  bool active = row < n;

  if (active)
    for (int k = lane; k < 256; k += 32) s_uc[w][k] = uc[(size_t)row * 256 + k];
  __syncthreads();

  float a0 = 0.f, a1 = 0.f;
  int j0 = lane, j1 = lane + 32;  // j1 valid for lanes 0..7
  if (active) {
    for (int k = 0; k < 256; k++) {
      float v = s_uc[w][k];
      a0 += v * Wxp[k * 40 + j0];
      if (j1 < 40) a1 += v * Wxp[k * 40 + j1];
    }
    s_dbc[w][j0] = a0;
    if (j1 < 40) s_dbc[w][j1] = a1;
  }
  __syncthreads();

  if (active) {
    for (int c = lane; c < 256; c += 32) {
      float acc = bdt[c];
#pragma unroll
      for (int r = 0; r < 8; r++) acc += s_dbc[w][r] * Wdt[r * 256 + c];
      float sp = (acc > 20.f) ? acc : log1pf(__expf(acc));
      delta[(size_t)row * 256 + c] = sp;
    }
    if (lane < 16) {
      Bm[(size_t)row * 16 + lane] = s_dbc[w][8 + lane];
      Cm[(size_t)row * 16 + lane] = s_dbc[w][24 + lane];
    }
  }
}

// ---------------- 7) scan pass A: per-chunk affine reduction ----------------
__global__ void k_scanA(const float* __restrict__ delta, const float* __restrict__ uc,
                        const float* __restrict__ Bm, const float* __restrict__ Alog,
                        float* __restrict__ P, float* __restrict__ Q, int n) {
  __shared__ float sB[CHUNK * 16];
  int c  = threadIdx.x;
  int g  = blockIdx.x;
  int t0 = g * CHUNK;
  // Async DMA the B tile (8KB) straight into LDS; tail garbage is never read
  // (inner loop bounded by L) and tail global reads stay inside the workspace.
  for (int e = threadIdx.x; e < (CHUNK * 16) / 4; e += blockDim.x)
    async_copy_b128(&sB[e * 4], Bm + (size_t)t0 * 16 + e * 4);
  async_wait0();
  __syncthreads();

  float Av[16];
#pragma unroll
  for (int s = 0; s < 16; s++) Av[s] = -__expf(Alog[c * 16 + s]);
  float p[16], q[16];
#pragma unroll
  for (int s = 0; s < 16; s++) { p[s] = 1.f; q[s] = 0.f; }

  int L = min(CHUNK, n - t0);
  for (int i = 0; i < L; i++) {
    int t = t0 + i;
    __builtin_prefetch(delta + (size_t)(t + 8) * 256 + c, 0, 0);
    __builtin_prefetch(uc + (size_t)(t + 8) * 256 + c, 0, 0);
    float d  = delta[(size_t)t * 256 + c];
    float du = d * uc[(size_t)t * 256 + c];
#pragma unroll
    for (int s = 0; s < 16; s++) {
      float a = __expf(d * Av[s]);
      p[s] *= a;
      q[s] = a * q[s] + du * sB[i * 16 + s];
    }
  }
  size_t base = ((size_t)g * 256 + c) * 16;
#pragma unroll
  for (int s = 0; s < 16; s++) { P[base + s] = p[s]; Q[base + s] = q[s]; }
}

// ---------------- 8) scan pass B: inter-chunk scan ----------------
__global__ void k_scanB(const float* __restrict__ P, const float* __restrict__ Q,
                        float* __restrict__ H, int nch) {
  int id = blockIdx.x * blockDim.x + threadIdx.x;  // 0..4095 == c*16+s
  if (id >= 4096) return;
  float h = 0.f;
  for (int g = 0; g < nch; g++) {
    size_t o = (size_t)g * 4096 + id;
    H[o] = h;
    h = P[o] * h + Q[o];
  }
}

// ---------------- 9) scan pass C: replay, fuse y=(ys+uc*D)*silu(z) -> f16 ----------------
__global__ void k_scanC(const float* __restrict__ delta, const float* __restrict__ uc,
                        const float* __restrict__ Bm, const float* __restrict__ Cm,
                        const float* __restrict__ H, const float* __restrict__ Alog,
                        const float* __restrict__ Dp, const float* __restrict__ xz,
                        _Float16* __restrict__ Y, int n) {
  __shared__ float sB[CHUNK * 16];
  __shared__ float sC[CHUNK * 16];
  int c  = threadIdx.x;
  int g  = blockIdx.x;
  int t0 = g * CHUNK;
  for (int e = threadIdx.x; e < (CHUNK * 16) / 4; e += blockDim.x) {
    async_copy_b128(&sB[e * 4], Bm + (size_t)t0 * 16 + e * 4);
    async_copy_b128(&sC[e * 4], Cm + (size_t)t0 * 16 + e * 4);
  }
  async_wait0();
  __syncthreads();

  float Av[16], h[16];
#pragma unroll
  for (int s = 0; s < 16; s++) {
    Av[s] = -__expf(Alog[c * 16 + s]);
    h[s]  = H[(size_t)g * 4096 + c * 16 + s];
  }
  float dc = Dp[c];

  int L = min(CHUNK, n - t0);
  for (int i = 0; i < L; i++) {
    int t = t0 + i;
    __builtin_prefetch(delta + (size_t)(t + 8) * 256 + c, 0, 0);
    __builtin_prefetch(uc + (size_t)(t + 8) * 256 + c, 0, 0);
    float d  = delta[(size_t)t * 256 + c];
    float uv = uc[(size_t)t * 256 + c];
    float du = d * uv;
    float zv = xz[(size_t)t * 512 + 256 + c];
    float y  = 0.f;
#pragma unroll
    for (int s = 0; s < 16; s++) {
      float a = __expf(d * Av[s]);
      h[s] = a * h[s] + du * sB[i * 16 + s];
      y += h[s] * sC[i * 16 + s];
    }
    y = y + uv * dc;
    y *= silu_f(zv);
    Y[(size_t)t * 256 + c] = (_Float16)y;
  }
}

// ---------------- 10) out = y @ W_out (N x 256 @ 256 x 128), f16 WMMA + scatter ----------------
__global__ void k_gemm_out(const _Float16* __restrict__ A /* N x 256 */,
                           const _Float16* __restrict__ Bt /* 128 x 256, K-contig */,
                           const int* __restrict__ idx,
                           float* __restrict__ out, int n) {
  int lane = threadIdx.x & 31;
  int w    = threadIdx.x >> 5;
  int sub = w >> 1, cg = w & 1;
  int row0 = blockIdx.x * 64 + sub * 16;
  int col0 = cg * 64;
  int m   = lane & 15;
  int kbA = (lane >> 4) << 3;
  int kbB = (lane >> 4) << 3;

  v8f zero = {0.f, 0.f, 0.f, 0.f, 0.f, 0.f, 0.f, 0.f};
  v8f acc[4] = {zero, zero, zero, zero};
  v8h zh = {0, 0, 0, 0, 0, 0, 0, 0};

  int r = row0 + m;
#pragma unroll
  for (int k0 = 0; k0 < 256; k0 += 32) {
    frag16 a;
    if (r < n) {
      const _Float16* ap = A + (size_t)r * 256 + k0 + kbA;
      a.h[0] = *(const v8h*)(ap);
      a.h[1] = *(const v8h*)(ap + 16);
    } else {
      a.h[0] = zh; a.h[1] = zh;
    }
#pragma unroll
    for (int g = 0; g < 4; ++g) {
      const _Float16* bp = Bt + (size_t)(col0 + g * 16 + m) * 256 + k0 + kbB;
      frag16 b;
      b.h[0] = *(const v8h*)(bp);
      b.h[1] = *(const v8h*)(bp + 16);
      acc[g] = wmma_f16(a.v, b.v, acc[g]);
    }
  }
  int rbase = row0 + ((lane >> 4) << 3);
#pragma unroll
  for (int v = 0; v < 8; v++) {
    int rr = rbase + v;
    if (rr >= n) continue;
    int orow = idx[rr];  // out[idx[r]] = out_sorted[r]
#pragma unroll
    for (int g = 0; g < 4; g++) {
      int cc = col0 + g * 16 + m;
      out[(size_t)orow * 128 + cc] = acc[g][v];
    }
  }
}

extern "C" void kernel_launch(void* const* d_in, const int* in_sizes, int n_in,
                              void* d_out, int out_size, void* d_ws, size_t ws_size,
                              hipStream_t stream) {
  const float* x     = (const float*)d_in[0];
  const float* Wsc   = (const float*)d_in[1];
  const float* bsc   = (const float*)d_in[2];
  const float* gamma = (const float*)d_in[3];
  const float* beta  = (const float*)d_in[4];
  const float* Win   = (const float*)d_in[5];
  const float* convw = (const float*)d_in[6];
  const float* convb = (const float*)d_in[7];
  const float* Wxp   = (const float*)d_in[8];
  const float* Wdt   = (const float*)d_in[9];
  const float* bdt   = (const float*)d_in[10];
  const float* Alog  = (const float*)d_in[11];
  const float* Dp    = (const float*)d_in[12];
  const float* Wout  = (const float*)d_in[13];
  float* out = (float*)d_out;

  const int n = in_sizes[0] / 128;  // N rows

  char* ws = (char*)d_ws;
  size_t off = 0;
  auto alloc = [&](size_t bytes) -> void* {
    void* p = ws + off;
    off = (off + bytes + 255) & ~(size_t)255;
    return p;
  };
  float*     scores = (float*)alloc((size_t)n * 4);
  int*       idx    = (int*)alloc((size_t)n * 4);
  _Float16*  xs_h   = (_Float16*)alloc((size_t)n * 128 * 2);
  float*     xz     = (float*)alloc((size_t)n * 512 * 4);
  float*     uc     = (float*)alloc((size_t)n * 256 * 4);
  float*     delta  = (float*)alloc((size_t)n * 256 * 4);
  float*     Bm     = (float*)alloc((size_t)n * 16 * 4);
  float*     Cm     = (float*)alloc((size_t)n * 16 * 4);
  _Float16*  y_h    = (_Float16*)alloc((size_t)n * 256 * 2);
  _Float16*  wt_in  = (_Float16*)alloc((size_t)512 * 128 * 2);
  _Float16*  wt_out = (_Float16*)alloc((size_t)128 * 256 * 2);
  int nch = (n + CHUNK - 1) / CHUNK;
  float* P = (float*)alloc((size_t)nch * 4096 * 4);
  float* Q = (float*)alloc((size_t)nch * 4096 * 4);
  float* H = (float*)alloc((size_t)nch * 4096 * 4);

  k_cvt_in<<<256, 256, 0, stream>>>(Win, wt_in);
  k_cvt_out<<<128, 256, 0, stream>>>(Wout, wt_out);
  k_score<<<(n + 7) / 8, 256, 0, stream>>>(x, Wsc, bsc, scores, n);
  k_rank<<<(n + 255) / 256, 256, 0, stream>>>(scores, idx, n);
  k_ln<<<(n + 7) / 8, 256, 0, stream>>>(x, idx, gamma, beta, xs_h, n);
  k_gemm_in<<<(n + 15) / 16, 256, 0, stream>>>(xs_h, wt_in, xz, n);
  k_conv<<<(n * 256 + 255) / 256, 256, 0, stream>>>(xz, convw, convb, uc, n);
  k_proj<<<(n + 7) / 8, 256, 0, stream>>>(uc, Wxp, Wdt, bdt, delta, Bm, Cm, n);
  k_scanA<<<nch, 256, 0, stream>>>(delta, uc, Bm, Alog, P, Q, n);
  k_scanB<<<16, 256, 0, stream>>>(P, Q, H, nch);
  k_scanC<<<nch, 256, 0, stream>>>(delta, uc, Bm, Cm, H, Alog, Dp, xz, y_h, n);
  k_gemm_out<<<(n + 63) / 64, 256, 0, stream>>>(y_h, wt_out, idx, out, n);
}
